// LeNet5Quantized_70334384439543
// MI455X (gfx1250) — compile-verified, matrix operand order checked by reading, not run on hardware
//
#include <hip/hip_runtime.h>
#include <stdint.h>

typedef __attribute__((ext_vector_type(16))) _Float16 v16h;
typedef __attribute__((ext_vector_type(8)))  _Float16 h8;
typedef __attribute__((ext_vector_type(8)))  float    v8f;

// quantize helpers (clip -> RNE round, matches jnp.round after clip)
__device__ __forceinline__ float qwv(float w, float invsw) {       // weight codes [-7,7]
  return rintf(fminf(fmaxf(w * invsw, -7.f), 7.f));
}
__device__ __forceinline__ float qact(float h, float qs) {         // relu+quant codes [0,15]
  return rintf(fminf(fmaxf(h * qs, 0.f), 15.f));                   // med3 + rndne
}
__device__ __forceinline__ int kpair(int j) { return (j < 4) ? 2*j : 2*j + 8; }

// fragment = two contiguous, 16B-aligned 8-half groups (K {s..s+7}, {s+16..s+23})
__device__ __forceinline__ v16h ld_frag(const _Float16* p0, const _Float16* p1) {
  h8 lo = *(const h8*)p0;
  h8 hi = *(const h8*)p1;
  return __builtin_shufflevector(lo, hi, 0,1,2,3,4,5,6,7,8,9,10,11,12,13,14,15);
}
// quantize 8 consecutive global f32 weights into an 8-half group
__device__ __forceinline__ h8 qw8(const float* __restrict__ wrow, int s, bool valid, float invsw) {
  h8 r;
  #pragma unroll
  for (int i = 0; i < 8; ++i) r[i] = (_Float16)0.f;
  if (valid) {
    const float4* p = (const float4*)(wrow + s);
    float4 a = p[0], b = p[1];
    float v[8] = {a.x, a.y, a.z, a.w, b.x, b.y, b.z, b.w};
    #pragma unroll
    for (int i = 0; i < 8; ++i) r[i] = (_Float16)qwv(v[i], invsw);
  }
  return r;
}
__device__ __forceinline__ v8f wmma16(const v16h& a, const v16h& b, v8f c) {
  return __builtin_amdgcn_wmma_f32_16x16x32_f16(false, a, false, b, (short)0, c, false, false);
}

// ================= front: quant -> conv0 -> pool -> conv1 -> pool =================
// One workgroup (8 wave32) per image; all intermediates in LDS.
__global__ void __launch_bounds__(256)
k_front(const float* __restrict__ x, const float* __restrict__ w0,
        const float* __restrict__ w1,
        const float* __restrict__ ps_in, const float* __restrict__ psw0,
        const float* __restrict__ psw1, const float* __restrict__ psa0,
        const float* __restrict__ psa1, _Float16* __restrict__ act1, int B)
{
  __shared__ __align__(16) _Float16 simg [34 * 32];       // quant input + 2 zero pad rows
  __shared__ __align__(16) _Float16 swgt [16 * 32];       // conv0 W [N=16][K=32]
  __shared__ __align__(16) _Float16 sW1  [16 * 320];      // conv1 W [N=16][K=320] padded
  __shared__ __align__(16) _Float16 sconv[784 * 8];       // conv0 codes, ch padded to 8
  __shared__ __align__(16) _Float16 sact0[14 * 24 * 8];   // pooled codes, x-stride 24
  __shared__ __align__(16) _Float16 sC   [100 * 16];      // conv1 pre-pool codes
  const int t = threadIdx.x;
  const int b = blockIdx.x;
  const float s_in = ps_in[0], sw0 = psw0[0], sw1 = psw1[0];
  const float sa0  = psa0[0],  sa1 = psa1[0];
  const float inv_sin = 1.0f / s_in;
  const float inv_sw0 = 1.0f / sw0;
  const float inv_sw1 = 1.0f / sw1;
  const float qs0 = (s_in * sw0) / sa0;                   // combined dequant->requant
  const float qs1 = (sa0  * sw1) / sa1;

  {  // stage + quantize input image (b128 loads); prefetch next image
    if (b + 1 < B) __builtin_prefetch(x + (size_t)(b + 1) * 1024);
    const float4 v = ((const float4*)(x + (size_t)b * 1024))[t];
    simg[t*4+0] = (_Float16)rintf(fminf(fmaxf(v.x * inv_sin, -128.f), 127.f));
    simg[t*4+1] = (_Float16)rintf(fminf(fmaxf(v.y * inv_sin, -128.f), 127.f));
    simg[t*4+2] = (_Float16)rintf(fminf(fmaxf(v.z * inv_sin, -128.f), 127.f));
    simg[t*4+3] = (_Float16)rintf(fminf(fmaxf(v.w * inv_sin, -128.f), 127.f));
  }
  if (t < 64) simg[1024 + t] = (_Float16)0.f;             // zero pad rows 32..33
  for (int i = t; i < 512; i += 256) {                    // conv0 weights [n][k]
    int n = i >> 5, k = i & 31;
    float q = (k < 25 && n < 6) ? qwv(w0[n*25 + k], inv_sw0) : 0.f;
    swgt[i] = (_Float16)q;
  }
  for (int i = t; i < 16*320; i += 256) {                 // conv1 weights [n][K=dy*64+dx*8+c]
    int n = i / 320, k = i % 320;
    int dy = k >> 6, dx = (k >> 3) & 7, cin = k & 7;
    float q = 0.f;
    if (dx < 5 && cin < 6) q = qwv(w1[(n*6 + cin)*25 + dy*5 + dx], inv_sw1);
    sW1[i] = (_Float16)q;
  }
  __syncthreads();

  const int wv = t >> 5, lane = t & 31;
  const int kb = (lane >> 4) << 3;
  const int nn = lane & 15;
  const int rbase = (lane >> 4) << 3;                     // C row offset per half-wave

  // ---- conv0: im2col K=32, k -> (dy=k/5, dx=k%5); taps 25..31 hit zero rows ----
  {
    const v16h fb = ld_frag(&swgt[nn*32 + kb], &swgt[nn*32 + kb + 16]);
    int offs[16];                                         // hoisted im2col offsets
    #pragma unroll
    for (int j = 0; j < 8; ++j)
      #pragma unroll
      for (int e = 0; e < 2; ++e) {
        int k = kb + kpair(j) + e;
        offs[2*j+e] = (k / 5) * 32 + (k % 5);
      }
    for (int tile = wv; tile < 49; tile += 8) {           // 784 outputs / 16
      int m = tile*16 + (lane & 15);
      int base = (m / 28) * 32 + (m % 28);
      v16h fa;
      #pragma unroll
      for (int i = 0; i < 16; ++i) fa[i] = simg[base + offs[i]];
      v8f c = {};
      c = wmma16(fa, fb, c);
      if (nn < 6) {
        #pragma unroll
        for (int r = 0; r < 8; ++r)
          sconv[(tile*16 + r + rbase)*8 + nn] = (_Float16)qact(c[r], qs0);
      }
    }
  }
  __syncthreads();

  // ---- 2x2 maxpool straight into the padded conv1 im2col layout ----
  for (int i = t; i < 14*24*8; i += 256) {
    int y = i / 192, rem = i % 192, xx = rem >> 3, cc = rem & 7;
    _Float16 v = (_Float16)0.f;
    if (xx < 14 && cc < 6) {
      int p0 = ((2*y)*28 + 2*xx)*8 + cc;
      _Float16 a0 = sconv[p0],       a1 = sconv[p0 + 8];
      _Float16 a2 = sconv[p0 + 224], a3 = sconv[p0 + 232];
      _Float16 m0 = a0 > a1 ? a0 : a1;
      _Float16 m1 = a2 > a3 ? a2 : a3;
      v = m0 > m1 ? m0 : m1;
    }
    sact0[i] = v;
  }
  __syncthreads();

  // ---- conv1: padded im2col K=320 (k = dy*64 + dx*8 + c) ----
  for (int tile = wv; tile < 7; tile += 8) {              // 100 outputs -> 7 tiles
    int m = tile*16 + (lane & 15);
    int mc = m < 100 ? m : 99;
    int base = ((mc / 10) * 24 + (mc % 10)) * 8;
    v8f c = {};
    for (int kk = 0; kk < 10; ++kk) {
      int s0 = kk*32 + kb, s1 = s0 + 16;
      int a0 = base + (s0 >> 6) * 192 + (s0 & 63);
      int a1 = base + (s1 >> 6) * 192 + (s1 & 63);
      v16h fa = ld_frag(&sact0[a0], &sact0[a1]);          // 2x ds_load_b128
      v16h fb = ld_frag(&sW1[nn*320 + s0], &sW1[nn*320 + s1]);
      c = wmma16(fa, fb, c);
    }
    #pragma unroll
    for (int r = 0; r < 8; ++r) {
      int m2 = tile*16 + r + rbase;
      if (m2 < 100) sC[m2*16 + nn] = (_Float16)qact(c[r], qs1);
    }
  }
  __syncthreads();

  // ---- pool -> act1 f16 [b][16][5][5] (channel-major: matches w2 flattening) ----
  for (int i = t; i < 400; i += 256) {
    int cc = i / 25, sp = i % 25, yy = sp / 5, xx = sp % 5;
    int p0 = ((2*yy)*10 + 2*xx)*16 + cc;
    _Float16 a0 = sC[p0],       a1 = sC[p0 + 16];
    _Float16 a2 = sC[p0 + 160], a3 = sC[p0 + 176];
    _Float16 m0 = a0 > a1 ? a0 : a1;
    _Float16 m1 = a2 > a3 ? a2 : a3;
    act1[(size_t)b*400 + i] = m0 > m1 ? m0 : m1;
  }
}

// ================= tail: conv2 -> fc0 -> fc1, 16 batch rows per WG =================
// conv2: M=16,N=120(pad128),K=400(pad416); fc0: N=84(pad96),K=120(pad128);
// fc1: N=10,K=84(pad96). Activations flow through LDS; pads stored as zero codes.
__global__ void __launch_bounds__(256)
k_tail(const _Float16* __restrict__ act1, const float* __restrict__ w2,
       const float* __restrict__ wf0, const float* __restrict__ wf1,
       const float* __restrict__ psw2, const float* __restrict__ psw3,
       const float* __restrict__ psw4, const float* __restrict__ psa1,
       const float* __restrict__ psa2, const float* __restrict__ psa3,
       float* __restrict__ out)
{
  __shared__ __align__(16) _Float16 sA [16 * 416];        // act1 rows (K padded)
  __shared__ __align__(16) _Float16 sH2[16 * 128];        // act2 codes (N padded)
  __shared__ __align__(16) _Float16 sH3[16 * 96];         // act3 codes (N padded)
  const int t = threadIdx.x;
  const int b0 = blockIdx.x * 16;
  const float sw2 = psw2[0], sw3 = psw3[0], sw4 = psw4[0];
  const float sa1 = psa1[0], sa2 = psa2[0], sa3 = psa3[0];
  const float inv_sw2 = 1.0f / sw2, inv_sw3 = 1.0f / sw3, inv_sw4 = 1.0f / sw4;
  const float qs2 = (sa1 * sw2) / sa2;
  const float qs3 = (sa2 * sw3) / sa3;
  const float fs  = sa3 * sw4;

  for (int i = t; i < 16 * 208; i += 256) {               // dword staging of A
    int m = i / 208, kd = i % 208;
    uint32_t v = 0;
    if (kd < 200) v = ((const uint32_t*)(act1 + (size_t)(b0 + m)*400))[kd];
    ((uint32_t*)sA)[i] = v;
  }
  __syncthreads();

  const int wv = t >> 5, lane = t & 31;
  const int kb = (lane >> 4) << 3, nl = lane & 15;
  const int mr = lane & 15;
  const int rbase = (lane >> 4) << 3;

  // ---- conv2: wave = column tile ----
  {
    const int n = wv*16 + nl;
    const float* wrow = w2 + (size_t)(n < 120 ? n : 0) * 400;
    v8f c = {};
    for (int kk = 0; kk < 13; ++kk) {
      int s0 = kk*32 + kb, s1 = s0 + 16;
      v16h fa = ld_frag(&sA[mr*416 + s0], &sA[mr*416 + s1]);
      v16h fb = __builtin_shufflevector(
          qw8(wrow, s0, (s0 < 400) && (n < 120), inv_sw2),
          qw8(wrow, s1, (s1 < 400) && (n < 120), inv_sw2),
          0,1,2,3,4,5,6,7,8,9,10,11,12,13,14,15);
      c = wmma16(fa, fb, c);
    }
    #pragma unroll
    for (int r = 0; r < 8; ++r) {                         // pads written as 0 (avoid NaN)
      float q = (n < 120) ? qact(c[r], qs2) : 0.f;
      sH2[(r + rbase)*128 + n] = (_Float16)q;
    }
  }
  __syncthreads();

  // ---- fc0: waves 0..5 cover N=96 ----
  if (wv < 6) {
    const int n = wv*16 + nl;
    const float* wrow = wf0 + (size_t)(n < 84 ? n : 0) * 120;
    v8f c = {};
    for (int kk = 0; kk < 4; ++kk) {
      int s0 = kk*32 + kb, s1 = s0 + 16;
      v16h fa = ld_frag(&sH2[mr*128 + s0], &sH2[mr*128 + s1]);
      v16h fb = __builtin_shufflevector(
          qw8(wrow, s0, (s0 < 120) && (n < 84), inv_sw3),
          qw8(wrow, s1, (s1 < 120) && (n < 84), inv_sw3),
          0,1,2,3,4,5,6,7,8,9,10,11,12,13,14,15);
      c = wmma16(fa, fb, c);
    }
    #pragma unroll
    for (int r = 0; r < 8; ++r) {
      float q = (n < 84) ? qact(c[r], qs3) : 0.f;
      sH3[(r + rbase)*96 + n] = (_Float16)q;
    }
  }
  __syncthreads();

  // ---- fc1: single wave finishes 16x10 ----
  if (wv == 0) {
    v8f c = {};
    for (int kk = 0; kk < 3; ++kk) {
      int s0 = kk*32 + kb;
      v16h fa = ld_frag(&sH3[mr*96 + s0], &sH3[mr*96 + s0 + 16]);
      v16h fb;
      #pragma unroll
      for (int j = 0; j < 8; ++j)
        #pragma unroll
        for (int e = 0; e < 2; ++e) {
          int k = s0 + kpair(j) + e;
          float q = (k < 84 && nl < 10) ? qwv(wf1[nl*84 + k], inv_sw4) : 0.f;
          fb[2*j+e] = (_Float16)q;
        }
      c = wmma16(fa, fb, c);
    }
    if (nl < 10) {
      #pragma unroll
      for (int r = 0; r < 8; ++r)
        out[(size_t)(b0 + r + rbase)*10 + nl] = c[r] * fs;
    }
  }
}

extern "C" void kernel_launch(void* const* d_in, const int* in_sizes, int n_in,
                              void* d_out, int out_size, void* d_ws, size_t ws_size,
                              hipStream_t stream) {
  const float* x    = (const float*)d_in[0];
  const float* w0   = (const float*)d_in[1];
  const float* w1   = (const float*)d_in[2];
  const float* w2   = (const float*)d_in[3];
  const float* wf0  = (const float*)d_in[4];
  const float* wf1  = (const float*)d_in[5];
  const float* s_in = (const float*)d_in[6];
  const float* sw0  = (const float*)d_in[7];
  const float* sw1  = (const float*)d_in[8];
  const float* sw2  = (const float*)d_in[9];
  const float* sw3  = (const float*)d_in[10];
  const float* sw4  = (const float*)d_in[11];
  const float* sa0  = (const float*)d_in[12];
  const float* sa1  = (const float*)d_in[13];
  const float* sa2  = (const float*)d_in[14];
  const float* sa3  = (const float*)d_in[15];

  const int B = in_sizes[0] / 1024;                       // x is [B,1,32,32]

  _Float16* act1 = (_Float16*)d_ws;                       // [B,16,5,5] f16 codes

  k_front<<<B,      256, 0, stream>>>(x, w0, w1, s_in, sw0, sw1, sa0, sa1, act1, B);
  k_tail <<<B / 16, 256, 0, stream>>>(act1, w2, wf0, wf1, sw2, sw3, sw4,
                                      sa1, sa2, sa3, (float*)d_out);
}
